// AdaptiveEdgeGNN_31181462569269
// MI455X (gfx1250) — matrix operand order repaired and checked by reading, not compiled, not run on hardware
//
#include <hip/hip_runtime.h>
#include <hip/hip_bf16.h>
#include <math.h>

// ---------------------------------------------------------------------------
// AdaptiveEdgeGNN on MI455X (gfx1250, wave32, WMMA).
// Heavy op: edge MLP layer-1 GEMM [E,128]@[128,64] via
// v_wmma_f32_16x16x32_bf16. The 16KB weight matrix lives in LDS (320KB/WGP);
// an opaque per-iteration offset stops LICM from hoisting the ds_loads into
// registers (which caused scratch spills in earlier rounds). Node tables are
// L2-resident (25.6MB f32 + 12.8MB bf16 << 192MB L2).
// ---------------------------------------------------------------------------

typedef __bf16 bf16_t;
typedef __attribute__((ext_vector_type(16))) __bf16 v16bf;
typedef __attribute__((ext_vector_type(8)))  __bf16 v8bf;
typedef __attribute__((ext_vector_type(8)))  float  v8f;

union V16 { v16bf v; v8bf h[2]; };

// ---------------- small prep kernels ----------------------------------------

__global__ void zero_f32(float* __restrict__ p, long long n) {
  long long i = (long long)blockIdx.x * blockDim.x + threadIdx.x;
  if (i < n) p[i] = 0.0f;
}

// transpose [K][Nn] f32 -> [Nn][K] bf16
__global__ void wprep_t(const float* __restrict__ w, bf16_t* __restrict__ wt,
                        int K, int Nn) {
  int i = blockIdx.x * blockDim.x + threadIdx.x;
  if (i >= K * Nn) return;
  int k = i / Nn, n = i % Nn;
  wt[n * K + k] = (bf16_t)w[i];
}

// conv1 node linear: h = x[:,0] * w[64] + b[64]  (in f32 and bf16)
__global__ void node_lin1(const float* __restrict__ x, const float* __restrict__ w,
                          const float* __restrict__ b, float* __restrict__ hF,
                          bf16_t* __restrict__ hB, long long n64) {
  long long i = (long long)blockIdx.x * blockDim.x + threadIdx.x;
  if (i >= n64) return;
  int f = (int)(i & 63);
  long long nd = i >> 6;
  float v = x[nd] * w[f] + b[f];
  hF[i] = v;
  hB[i] = (bf16_t)v;
}

// ---------------- edge kernel (the 52-GFLOP part) ----------------------------
// Persistent waves; one wave handles a stream of 16-edge tiles:
//   ef = [h_bf16[src], h_bf16[dst]]  (16 x 128)
//   L1 = relu(ef @ w1 + b1)          -> 16 WMMA bf16 ops (4 Kblk x 4 Nblk)
//   z  = L1 @ w2 + b2 ; g = sigmoid(z)
//   atomicAdd(agg[dst], g * h_f32[src])
// Weights live in LDS; B fragments are re-read per tile via ds_load (the
// opaque offset prevents hoist-then-spill), one N-block (32 VGPRs) at a time.
__global__ void edge_conv_kernel(const float* __restrict__ hf,   // [N,64] f32
                                 const bf16_t* __restrict__ hb,  // [N,64] bf16
                                 const int* __restrict__ src,    // [E]
                                 const int* __restrict__ dst,    // [E]
                                 const bf16_t* __restrict__ w1t, // [64][128] bf16 (m1w^T)
                                 const float* __restrict__ b1,   // [64]
                                 const float* __restrict__ w2,   // [64]  (m2w)
                                 const float* __restrict__ b2s,  // [1]
                                 float* __restrict__ agg,        // [N,64] accum
                                 int nEdges) {
  __shared__ bf16_t sW[64 * 128];       // 16 KB, w1t block-resident

  // cooperative LDS fill (u32 granularity), once per block
  {
    const unsigned* gw = (const unsigned*)w1t;
    unsigned* sw = (unsigned*)sW;
    for (int i = threadIdx.x; i < 64 * 128 / 2; i += blockDim.x) sw[i] = gw[i];
  }
  __syncthreads();

  const int  lane   = threadIdx.x & 31;
  const int  wave   = threadIdx.x >> 5;
  const int  el     = lane & 15;
  const int  nIdx   = lane & 15;
  const bool hiHalf = lane >= 16;

  float biasv[4], w2v[4];
#pragma unroll
  for (int nb = 0; nb < 4; ++nb) {
    biasv[nb] = b1[nb * 16 + nIdx];
    w2v[nb]   = w2[nb * 16 + nIdx];
  }
  const float b2 = b2s[0];

  const int nTiles = (nEdges + 15) / 16;
  const int nWaves = gridDim.x * (blockDim.x >> 5);

  for (int tile = blockIdx.x * (blockDim.x >> 5) + wave; tile < nTiles;
       tile += nWaves) {
    // Opaque zero: redefined by volatile asm every iteration so the LDS
    // B-fragment loads below are loop-variant and cannot be LICM-hoisted
    // (hoisting them spills 128 VGPRs of weights to scratch).
    unsigned zoff = 0;
    asm volatile("" : "+v"(zoff));

    int e = tile * 16 + el;
    if (e >= nEdges) e = nEdges - 1;    // E is a multiple of 16; safety only
    const int s = src[e];
    const int d = dst[e];

    // prefetch f32 src row (used by the scatter) while WMMAs run
    __builtin_prefetch(hf + (size_t)s * 64, 0, 0);

    // A fragments: 16x32, contiguous k order (matches the B loads).
    // lanes 0-15 : M=lane,    K = 0..15 (+32*kb)
    // lanes 16-31: M=lane-16, K = 16..31 (+32*kb)
    // features 0..63 -> hb[src], 64..127 -> hb[dst]
    V16 A[4];
#pragma unroll
    for (int kb = 0; kb < 4; ++kb) {
      const int f0 = kb * 32 + (hiHalf ? 16 : 0);
      const bf16_t* rowa = (kb < 2) ? (hb + (size_t)s * 64)
                                    : (hb + (size_t)d * 64 - 64);
      A[kb].h[0] = *(const v8bf*)(rowa + f0);
      A[kb].h[1] = *(const v8bf*)(rowa + f0 + 8);
    }

    float p[8];
#pragma unroll
    for (int v = 0; v < 8; ++v) p[v] = 0.0f;

#pragma unroll
    for (int nb = 0; nb < 4; ++nb) {
      // B 32x16 layout: lanes 0-15 K=0..15 of column N=nIdx; lanes 16-31
      // K=16..31. One N-block of B at a time -> 32 VGPRs live, no spills.
      const bf16_t* wrow =
          sW + zoff + (size_t)(nb * 16 + nIdx) * 128 + (hiHalf ? 16 : 0);
      v8f c = {};
#pragma unroll
      for (int kb = 0; kb < 4; ++kb) {
        v16bf b = *(const v16bf*)(wrow + kb * 32);   // 2x ds_load_b128
        c = __builtin_amdgcn_wmma_f32_16x16x32_bf16(false, A[kb].v, false, b,
                                                    (short)0, c, false, false);
      }
#pragma unroll
      for (int v = 0; v < 8; ++v) {
        float t = c[v] + biasv[nb];     // C layout: row M=v(+8 hi), col nIdx
        t = t > 0.0f ? t : 0.0f;        // relu
        p[v] += t * w2v[nb];            // layer-2 dot partial
      }
    }

    // reduce across the 16 lanes of each half -> gates replicated per half
#pragma unroll
    for (int v = 0; v < 8; ++v) {
      float t = p[v];
      t += __shfl_xor(t, 1);
      t += __shfl_xor(t, 2);
      t += __shfl_xor(t, 4);
      t += __shfl_xor(t, 8);
      p[v] = t;
    }

    // scatter: msg = gate * h_f32[src], atomic add at dst (agg L2-resident)
#pragma unroll
    for (int ee = 0; ee < 16; ++ee) {
      float z  = __shfl(p[ee & 7], (ee & 8) ? 16 : 0);
      float g  = 1.0f / (1.0f + __expf(-(z + b2)));
      int   se = __shfl(s, ee);
      int   de = __shfl(d, ee);
      float m0 = g * hf[(size_t)se * 64 + lane];
      float m1 = g * hf[(size_t)se * 64 + lane + 32];
      atomicAdd(&agg[(size_t)de * 64 + lane], m0);
      atomicAdd(&agg[(size_t)de * 64 + lane + 32], m1);
    }
  }
}

// ---------------- conv2 node linear: h2 = relu(agg1) @ c2_lw + b ------------
// [N,64]@[64,64] via WMMA: persistent waves, 16 node rows per tile,
// 2 Kblk x 4 Nblk, weights LDS-resident (8 KB), anti-hoist opaque offset.
__global__ void node_gemm_kernel(const float* __restrict__ aggIn, // [N,64]
                                 const bf16_t* __restrict__ wt,   // [64][64] c2_lw^T
                                 const float* __restrict__ bias,  // [64]
                                 float* __restrict__ hOutF,       // [N,64]
                                 bf16_t* __restrict__ hOutB,      // [N,64]
                                 int nNodes) {
  __shared__ bf16_t sW[64 * 64];        // 8 KB

  {
    const unsigned* gw = (const unsigned*)wt;
    unsigned* sw = (unsigned*)sW;
    for (int i = threadIdx.x; i < 64 * 64 / 2; i += blockDim.x) sw[i] = gw[i];
  }
  __syncthreads();

  const int  lane   = threadIdx.x & 31;
  const int  wave   = threadIdx.x >> 5;
  const int  el     = lane & 15;
  const bool hiHalf = lane >= 16;

  float biasv[4];
#pragma unroll
  for (int nb = 0; nb < 4; ++nb) biasv[nb] = bias[nb * 16 + el];

  const int nTiles = (nNodes + 15) / 16;
  const int nWaves = gridDim.x * (blockDim.x >> 5);

  for (int tile = blockIdx.x * (blockDim.x >> 5) + wave; tile < nTiles;
       tile += nWaves) {
    unsigned zoff = 0;
    asm volatile("" : "+v"(zoff));      // anti-LICM (see edge kernel)

    int row = tile * 16 + el;
    if (row >= nNodes) row = nNodes - 1;  // N multiple of 16; safety only

    V16 A[2];
#pragma unroll
    for (int kb = 0; kb < 2; ++kb) {
      const int f0 = kb * 32 + (hiHalf ? 16 : 0);
#pragma unroll
      for (int h = 0; h < 2; ++h) {
        v8f a = *(const v8f*)(aggIn + (size_t)row * 64 + f0 + h * 8);
        v8bf o;
#pragma unroll
        for (int j = 0; j < 8; ++j) {
          float t = a[j];
          t = t > 0.0f ? t : 0.0f;      // relu fused on the A load
          o[j] = (bf16_t)t;
        }
        A[kb].h[h] = o;
      }
    }

#pragma unroll
    for (int nb = 0; nb < 4; ++nb) {
      const bf16_t* wrow =
          sW + zoff + (size_t)(nb * 16 + el) * 64 + (hiHalf ? 16 : 0);
      v8f c = {};
#pragma unroll
      for (int kb = 0; kb < 2; ++kb) {
        v16bf b = *(const v16bf*)(wrow + kb * 32);
        c = __builtin_amdgcn_wmma_f32_16x16x32_bf16(false, A[kb].v, false, b,
                                                    (short)0, c, false, false);
      }
#pragma unroll
      for (int v = 0; v < 8; ++v) {
        float val = c[v] + biasv[nb];
        int   m   = v + (hiHalf ? 8 : 0);
        long long r = (long long)tile * 16 + m;
        if (r < nNodes) {
          size_t o = (size_t)r * 64 + nb * 16 + el;
          hOutF[o] = val;
          hOutB[o] = (bf16_t)val;
        }
      }
    }
  }
}

// ---------------- pooling + head --------------------------------------------

__global__ void pool_kernel(const float* __restrict__ agg2,
                            const int* __restrict__ batch,
                            float* __restrict__ gsum, float* __restrict__ gcnt,
                            long long n64) {
  long long i = (long long)blockIdx.x * blockDim.x + threadIdx.x;
  if (i >= n64) return;
  int f = (int)(i & 63);
  long long nd = i >> 6;
  int g = batch[nd];
  float v = agg2[i];
  v = v > 0.0f ? v : 0.0f;             // relu after conv2
  atomicAdd(&gsum[(size_t)g * 64 + f], v);
  if (f == 0) atomicAdd(&gcnt[g], 1.0f);
}

__global__ void head_kernel(const float* __restrict__ gsum,
                            const float* __restrict__ gcnt,
                            const float* __restrict__ hw,
                            const float* __restrict__ hb,
                            float* __restrict__ out, int G) {
  const int lane = threadIdx.x & 31;
  const int wave = threadIdx.x >> 5;
  const int g    = blockIdx.x * (blockDim.x >> 5) + wave;
  if (g >= G) return;
  float cnt = gcnt[g];
  cnt = cnt > 1.0f ? cnt : 1.0f;
  float acc = gsum[(size_t)g * 64 + lane] * hw[lane] +
              gsum[(size_t)g * 64 + lane + 32] * hw[lane + 32];
  acc /= cnt;
  acc += __shfl_xor(acc, 16);
  acc += __shfl_xor(acc, 8);
  acc += __shfl_xor(acc, 4);
  acc += __shfl_xor(acc, 2);
  acc += __shfl_xor(acc, 1);
  if (lane == 0) out[g] = acc + hb[0];
}

// ---------------- launcher ---------------------------------------------------

extern "C" void kernel_launch(void* const* d_in, const int* in_sizes, int n_in,
                              void* d_out, int out_size, void* d_ws,
                              size_t ws_size, hipStream_t stream) {
  const float* x        = (const float*)d_in[0];
  const int*   eidx     = (const int*)d_in[1];
  const int*   batch    = (const int*)d_in[2];
  const float* c1_lw    = (const float*)d_in[3];
  const float* c1_lb    = (const float*)d_in[4];
  const float* c1_m1w   = (const float*)d_in[5];
  const float* c1_m1b   = (const float*)d_in[6];
  const float* c1_m2w   = (const float*)d_in[7];
  const float* c1_m2b   = (const float*)d_in[8];
  const float* c2_lw    = (const float*)d_in[9];
  const float* c2_lb    = (const float*)d_in[10];
  const float* c2_m1w   = (const float*)d_in[11];
  const float* c2_m1b   = (const float*)d_in[12];
  const float* c2_m2w   = (const float*)d_in[13];
  const float* c2_m2b   = (const float*)d_in[14];
  const float* head_w   = (const float*)d_in[15];
  const float* head_b   = (const float*)d_in[16];

  const int N = in_sizes[0];          // x is [N,1]
  const int E = in_sizes[1] / 2;      // edge_index is [2,E]
  const int G = out_size;             // out is [G,1]
  const int* src = eidx;
  const int* dst = eidx + E;

  // workspace carve-up (~64 MB)
  char*  ws  = (char*)d_ws;
  size_t off = 0;
  auto carve = [&](size_t bytes) -> void* {
    void* p = ws + off;
    off = (off + bytes + 255) & ~(size_t)255;
    return p;
  };
  const long long n64 = (long long)N * 64;
  float*  hF    = (float*) carve(n64 * 4);      // h (f32)
  bf16_t* hB    = (bf16_t*)carve(n64 * 2);      // h (bf16)
  float*  agg   = (float*) carve(n64 * 4);      // aggregation buffer (reused)
  bf16_t* w1tA  = (bf16_t*)carve(128 * 64 * 2); // c1_m1w^T bf16
  bf16_t* w1tB  = (bf16_t*)carve(128 * 64 * 2); // c2_m1w^T bf16
  bf16_t* w2t   = (bf16_t*)carve(64 * 64 * 2);  // c2_lw^T bf16
  float*  gsum  = (float*) carve((size_t)G * 64 * 4);
  float*  gcnt  = (float*) carve((size_t)G * 4);
  (void)ws_size;

  const int TPB = 256;
  const int wavesPerBlk = TPB / 32;

  // weight prep
  wprep_t<<<(128 * 64 + TPB - 1) / TPB, TPB, 0, stream>>>(c1_m1w, w1tA, 128, 64);
  wprep_t<<<(128 * 64 + TPB - 1) / TPB, TPB, 0, stream>>>(c2_m1w, w1tB, 128, 64);
  wprep_t<<<(64 * 64 + TPB - 1) / TPB, TPB, 0, stream>>>(c2_lw, w2t, 64, 64);

  // persistent grids: enough waves to saturate, few enough to amortize the
  // LDS weight fill over many tiles per block.
  int eTiles  = (E + 15) / 16;
  int eBlocks = (eTiles + wavesPerBlk - 1) / wavesPerBlk;
  if (eBlocks > 1536) eBlocks = 1536;
  int nTilesN  = (N + 15) / 16;
  int nBlocksN = (nTilesN + wavesPerBlk - 1) / wavesPerBlk;
  if (nBlocksN > 1024) nBlocksN = 1024;

  // ---- conv1 ----
  zero_f32<<<(int)((n64 + TPB - 1) / TPB), TPB, 0, stream>>>(agg, n64);
  node_lin1<<<(int)((n64 + TPB - 1) / TPB), TPB, 0, stream>>>(x, c1_lw, c1_lb,
                                                              hF, hB, n64);
  edge_conv_kernel<<<eBlocks, TPB, 0, stream>>>(hF, hB, src, dst, w1tA,
                                                c1_m1b, c1_m2w, c1_m2b, agg, E);

  // ---- conv2 node linear: h2 = relu(agg) @ c2_lw + b ----
  node_gemm_kernel<<<nBlocksN, TPB, 0, stream>>>(agg, w2t, c2_lb, hF, hB, N);

  // ---- conv2 edge pass (agg reused) ----
  zero_f32<<<(int)((n64 + TPB - 1) / TPB), TPB, 0, stream>>>(agg, n64);
  edge_conv_kernel<<<eBlocks, TPB, 0, stream>>>(hF, hB, src, dst, w1tB,
                                                c2_m1b, c2_m2w, c2_m2b, agg, E);

  // ---- pool + head ----
  zero_f32<<<(G * 64 + TPB - 1) / TPB, TPB, 0, stream>>>(gsum,
                                                         (long long)G * 64);
  zero_f32<<<1, TPB, 0, stream>>>(gcnt, G);
  pool_kernel<<<(int)((n64 + TPB - 1) / TPB), TPB, 0, stream>>>(agg, batch,
                                                                gsum, gcnt, n64);
  head_kernel<<<(G + wavesPerBlk - 1) / wavesPerBlk, TPB, 0, stream>>>(
      gsum, gcnt, head_w, head_b, (float*)d_out, G);
}
